// Layer_61022895342037
// MI455X (gfx1250) — matrix operand order repaired and checked by reading, not compiled, not run on hardware
//
#include <hip/hip_runtime.h>

typedef __attribute__((ext_vector_type(16))) __bf16 v16bf;
typedef __attribute__((ext_vector_type(8)))  float  v8f;
typedef __attribute__((ext_vector_type(4)))  int    v4i;

#define TAU_F    0.5f
#define THRESH_F 1.0f
#define BN_EPS_F 1e-5f

#define NB   32
#define NT   8
#define CIN  64
#define COUT 128
#define HH   32
#define WWD  32
#define NIMG (NB*NT)        // 256 flattened images
#define HWN  (HH*WWD)       // 1024
#define HP   34             // padded H (zero halo)
#define WP   34             // padded W (zero halo)
#define CPAIR 32            // 64 channels as 32 bf16-pairs
#define SLAB_CS 20          // cpair stride in slab (padded 16->20, 16B aligned, bank-spread)
#define SLABD (3*WP*SLAB_CS)   // 2040 dwords per buffer

// workspace layout (bytes)
//   y      : NIMG*COUT*HWN f32                = 134217728
//   xpack  : NIMG*HP*WP*CPAIR dwords          =  37879808  (bf16 pairs, zero halo)
//   apack  : 8*18*32*8 dwords                 =    147456  (WMMA-swizzled bf16 weights)
//   gstats : 256 f32 (sum|sumsq), params: 256 f32 (scale|shift)
#define XPACK_OFF 134217728ull
#define APACK_OFF (XPACK_OFF + 37871616ull)
#define STATS_OFF (APACK_OFF + 147456ull)

#if defined(__has_builtin)
#  if __has_builtin(__builtin_amdgcn_global_load_async_to_lds_b128)
#    define HAVE_ASYNC_LDS 1
#  else
#    define HAVE_ASYNC_LDS 0
#  endif
#  if __has_builtin(__builtin_amdgcn_s_wait_asynccnt)
#    define HAVE_WAIT_ASYNC 1
#  else
#    define HAVE_WAIT_ASYNC 0
#  endif
#else
#  define HAVE_ASYNC_LDS 0
#  define HAVE_WAIT_ASYNC 0
#endif

__device__ __forceinline__ void wait_async0() {
#if HAVE_WAIT_ASYNC
    __builtin_amdgcn_s_wait_asynccnt(0);
#else
    asm volatile("s_wait_asynccnt 0x0" ::: "memory");
#endif
}

__device__ __forceinline__ unsigned short f2bf(float f) {
    unsigned int u = __float_as_uint(f);
    u += 0x7FFFu + ((u >> 16) & 1u);          // round-to-nearest-even
    return (unsigned short)(u >> 16);
}

// ---------------------------------------------------------------------------
// Kernel 1: weights f32 -> bf16, pre-swizzled into the 16x32 A-matrix lane
// layout (m = lane%16, k = 16*(lane/16) + element). K order: s = chalf*9 + (ky*3+kx), k_local = cin%32.
// Also zeroes the BN stat accumulators (atomically accumulated per launch).
// ---------------------------------------------------------------------------
__global__ void pack_weights(const float* __restrict__ Wt,
                             unsigned int* __restrict__ apack,
                             float* __restrict__ gstats) {
    int gid = blockIdx.x * 256 + threadIdx.x;
    if (blockIdx.x == 0) gstats[threadIdx.x] = 0.0f;
    if (gid >= 8 * 18 * 32 * 8) return;
    int j    = gid & 7;          // VGPR dword index (2 bf16)
    int lane = (gid >> 3) & 31;
    int ts   = gid >> 8;         // tile*18 + s
    int s    = ts % 18;
    int tile = ts / 18;
    int m    = tile * 16 + (lane & 15);
    int klocal = ((lane >= 16) ? 16 : 0) + j * 2;
    int chalf = s / 9, r = s % 9;
    int ky = r / 3, kx = r % 3;
    int cin0 = chalf * 32 + klocal;
    unsigned short b0 = f2bf(Wt[((m * CIN + cin0    ) * 3 + ky) * 3 + kx]);
    unsigned short b1 = f2bf(Wt[((m * CIN + cin0 + 1) * 3 + ky) * 3 + kx]);
    apack[gid] = (unsigned int)b0 | ((unsigned int)b1 << 16);
}

// ---------------------------------------------------------------------------
// Kernel 2: x f32 [n][c][h][w] -> bf16 channel-pair dwords with zero halo:
// xpack[n][hp(34)][wp(34)][cpair(32)].  Removes all conversion and boundary
// logic from the conv hot loop and enables raw async DMA into LDS.
// ---------------------------------------------------------------------------
__global__ void pack_x(const float* __restrict__ x,
                       unsigned int* __restrict__ xpack) {
    size_t idx = (size_t)blockIdx.x * 256 + threadIdx.x;
    int cpair = (int)(idx & (CPAIR - 1));
    int wp = (int)((idx >> 5) % WP);
    int hp = (int)((idx >> 5) / WP % HP);
    int n  = (int)(idx / ((size_t)HP * WP * CPAIR));
    unsigned int dw = 0u;
    if (hp >= 1 && hp <= HH && wp >= 1 && wp <= WWD) {
        const float* p = x + ((size_t)(n * CIN + cpair * 2) * HH + (hp - 1)) * WWD + (wp - 1);
        dw = (unsigned int)f2bf(p[0]) | ((unsigned int)f2bf(p[HWN]) << 16);
    }
    xpack[idx] = dw;
}

// ---------------------------------------------------------------------------
// Kernel 3: implicit-GEMM 3x3 conv via v_wmma_f32_16x16x32_bf16.
// Block = 256 thr = 8 waves = all 128 couts x one full 32-wide image row.
// Each wave: 2 N-tiles sharing each A fragment. Input slab (32ch x 3rows x
// 34cols) double-buffered in LDS; next cin-half DMA'd asynchronously while
// WMMAs run on the current one.
// ---------------------------------------------------------------------------
__global__ void __launch_bounds__(256)
conv_wmma(const unsigned int* __restrict__ xpack,
          const unsigned int* __restrict__ apack,
          const float* __restrict__ bias, float* __restrict__ y,
          float* __restrict__ gstats) {
    __shared__ unsigned int slab[2][SLABD];   // [(row*34+col)*20 + cpair]
    __shared__ float chRed[256];              // sum[128] | sumsq[128]

    const int tid  = threadIdx.x;
    const int lane = tid & 31;
    const int wave = tid >> 5;                // cout tile 0..7
    const int h    = blockIdx.x;              // 0..31
    const int n    = blockIdx.y;              // 0..255
    const int nn   = lane & 15;
    const int hiK  = (lane >> 4) & 1;
    const int kq   = hiK * 8;

    union AU { v16bf v; unsigned int u[8]; };
    v8f acc[2] = {};

    // base of padded input window: rows h..h+2 (padded coords), cols 0..33
    const unsigned int* gbase = xpack + ((size_t)n * HP + h) * WP * CPAIR;

    auto fill = [&](int buf, int chalf) {
        // 102 (row,col) cells x 16 dwords = 408 x b128 units
        for (int u = tid; u < 3 * WP * 4; u += 256) {
            int q   = u & 3;
            int rc  = u >> 2;
            int col = rc % WP;
            int row = rc / WP;
            const unsigned int* g = gbase + ((size_t)row * WP + col) * CPAIR
                                          + chalf * 16 + q * 4;
            unsigned int* l = &slab[buf][(row * WP + col) * SLAB_CS + q * 4];
#if HAVE_ASYNC_LDS
            __builtin_amdgcn_global_load_async_to_lds_b128(
                (__attribute__((address_space(1))) v4i*)(v4i*)g,
                (__attribute__((address_space(3))) v4i*)(v4i*)l,
                0, 0);
#else
            *(uint4*)l = *(const uint4*)g;
#endif
        }
    };

    auto compute = [&](int buf, int chalf) {
        #pragma unroll
        for (int r = 0; r < 9; ++r) {
            const int s = chalf * 9 + r;
            AU a;
            const unsigned int* ap = apack + (((wave * 18 + s) * 32 + lane) << 3);
            #pragma unroll
            for (int q = 0; q < 8; ++q) a.u[q] = ap[q];   // 2x b128, L2-resident
            const int dy = r / 3, dx = r % 3;
            #pragma unroll
            for (int t = 0; t < 2; ++t) {                 // two 16-col tiles
                AU b;
                const unsigned int* bp =
                    &slab[buf][(dy * WP + t * 16 + nn + dx) * SLAB_CS + kq];
                #pragma unroll
                for (int v = 0; v < 8; ++v) b.u[v] = bp[v];   // 2x ds_load_b128
                acc[t] = __builtin_amdgcn_wmma_f32_16x16x32_bf16(
                    false, a.v, false, b.v, (short)0, acc[t], false, false);
            }
        }
    };

    fill(0, 0);
#if HAVE_ASYNC_LDS
    wait_async0();
#endif
    __syncthreads();
    fill(1, 1);            // prefetch cin-half 1 while computing cin-half 0
    compute(0, 0);
#if HAVE_ASYNC_LDS
    wait_async0();
#endif
    __syncthreads();
    compute(1, 1);

    // bias, BN partial stats, store conv output
    chRed[tid] = 0.0f;
    __syncthreads();
    #pragma unroll
    for (int rI = 0; rI < 8; ++rI) {
        const int cout = wave * 16 + hiK * 8 + rI;   // C/D layout: lanes>=16 -> M+8
        const float bi = bias[cout];
        #pragma unroll
        for (int t = 0; t < 2; ++t) {
            float val = acc[t][rI] + bi;
            atomicAdd(&chRed[cout],       val);          // ds_add_f32
            atomicAdd(&chRed[128 + cout], val * val);
            y[(((size_t)n * COUT + cout) * HH + h) * WWD + t * 16 + nn] = val;
        }
    }
    __syncthreads();
    atomicAdd(&gstats[tid], chRed[tid]);                 // one global add/channel
}

// ---------------------------------------------------------------------------
// Kernel 4: fold batch stats into per-channel scale/shift.
// ---------------------------------------------------------------------------
__global__ void bn_params(const float* __restrict__ gamma,
                          const float* __restrict__ beta,
                          const float* __restrict__ gstats,
                          float* __restrict__ params) {
    int c = threadIdx.x;
    if (c >= COUT) return;
    const float invN = 1.0f / (float)(NIMG * HWN);
    float mean = gstats[c] * invN;
    float var  = gstats[128 + c] * invN - mean * mean;
    float sc   = gamma[c] * rsqrtf(var + BN_EPS_F);
    params[c]       = sc;
    params[128 + c] = beta[c] - mean * sc;
}

// ---------------------------------------------------------------------------
// Kernel 5: BN affine + LIF scan over T (sequential in t), emit spikes.
// ---------------------------------------------------------------------------
__global__ void __launch_bounds__(256)
lif_scan(const float* __restrict__ y, const float* __restrict__ params,
         float* __restrict__ out) {
    size_t gid = (size_t)blockIdx.x * 256 + threadIdx.x;  // NB*COUT*HWN
    int hw = (int)(gid & (HWN - 1));
    int c  = (int)((gid >> 10) & (COUT - 1));
    int b  = (int)(gid >> 17);
    float sc = params[c], sh = params[128 + c];
    size_t base = ((size_t)b * NT * COUT + c) * HWN + hw;
    const size_t tstride = (size_t)COUT * HWN;
    float mem = 0.0f;
    #pragma unroll
    for (int t = 0; t < NT; ++t) {
        float yv = y[base + (size_t)t * tstride] * sc + sh;
        mem = mem * TAU_F + yv;
        float spike = (mem - THRESH_F > 0.0f) ? 1.0f : 0.0f;
        out[base + (size_t)t * tstride] = spike;
        mem = (1.0f - spike) * mem;
    }
}

// ---------------------------------------------------------------------------
extern "C" void kernel_launch(void* const* d_in, const int* in_sizes, int n_in,
                              void* d_out, int out_size, void* d_ws, size_t ws_size,
                              hipStream_t stream) {
    const float* x     = (const float*)d_in[0];
    const float* Wt    = (const float*)d_in[1];
    const float* bias  = (const float*)d_in[2];
    const float* gamma = (const float*)d_in[3];
    const float* beta  = (const float*)d_in[4];
    float* out = (float*)d_out;

    char* ws = (char*)d_ws;
    float*        y      = (float*)ws;
    unsigned int* xpack  = (unsigned int*)(ws + XPACK_OFF);
    unsigned int* apack  = (unsigned int*)(ws + APACK_OFF);
    float*        gstats = (float*)(ws + STATS_OFF);
    float*        params = gstats + 256;

    pack_weights<<<144, 256, 0, stream>>>(Wt, apack, gstats);
    pack_x<<<(NIMG * HP * WP * CPAIR) / 256, 256, 0, stream>>>(x, xpack);
    conv_wmma<<<dim3(HH, NIMG), 256, 0, stream>>>(xpack, apack, bias, y, gstats);
    bn_params<<<1, 128, 0, stream>>>(gamma, beta, gstats, params);
    lif_scan<<<(NB * COUT * HWN) / 256, 256, 0, stream>>>(y, params, out);
}
